// GraphConvolutionChebyshev_86466281603623
// MI455X (gfx1250) — matrix operand order repaired and checked by reading, not compiled, not run on hardware
//
#include <hip/hip_runtime.h>

// ---------------------------------------------------------------------------
// GraphConvolution on MI455X (gfx1250, wave32, WMMA + TDM)
//   out = segment_sum(vals * (x@W2)[cols] -> rows) + x@W1 + bias
// gemm_dual: 2500 blocks of 8 waves; block tile = 128 rows x 64 cols.
// Packed bf16 weights staged via TENSOR_LOAD_TO_LDS (one 32KB TDM copy per
// matrix per block), WMMA B fragments then served from LDS.
// ---------------------------------------------------------------------------

typedef __attribute__((ext_vector_type(16))) __bf16 v16bf;
typedef __attribute__((ext_vector_type(8)))  float  v8f;
typedef __attribute__((ext_vector_type(4)))  unsigned tdm_g0_t;
typedef __attribute__((ext_vector_type(8)))  int      tdm_g1_t;
typedef __attribute__((ext_vector_type(4)))  int      tdm_g2_t;
typedef __attribute__((ext_vector_type(8)))  int      tdm_g3_t;

#define BB 4
#define NN 20000
#define FF 256
#define OO 256
#define EE 160000
#define MROWS (BB * NN)          // 80000 flattened rows of [B,N,F]

static __device__ __forceinline__ unsigned short f2bf(float f) {
  unsigned u = __float_as_uint(f);
  unsigned rnd = ((u >> 16) & 1u) + 0x7fffu;   // round-to-nearest-even
  return (unsigned short)((u + rnd) >> 16);
}
static __device__ __forceinline__ unsigned pack2(float lo, float hi) {
  return (unsigned)f2bf(lo) | ((unsigned)f2bf(hi) << 16);
}

// ---------------------------------------------------------------------------
// Kernel 1: swizzle fp32 weights [256x256] into bf16 WMMA B-fragment layout.
// Fragment order per matrix: [nq(4)][ks(8)][nf(4)][lane(32)][8 dwords]
// so each N-quarter's working set (all K) is one contiguous 32 KB region.
// B packing used here (contiguous k, not the ISA table order): lanes 0-15: N=lane, K=0..15 (dword j -> 2j,2j+1)
//                            lanes 16-31: N=lane-16, K=16..31
// ---------------------------------------------------------------------------
__global__ __launch_bounds__(256)
void pack_weights(const float* __restrict__ w1, const float* __restrict__ w2,
                  unsigned* __restrict__ p1, unsigned* __restrict__ p2) {
  int gid  = blockIdx.x * 256 + threadIdx.x;   // 0..8191
  int mat  = gid >> 12;                        // 0: w1, 1: w2
  int rem  = gid & 4095;
  int f    = rem >> 5;                         // 0..127 = ((nq*8+ks)*4+nf)
  int lane = rem & 31;
  int nq   = f >> 5;
  int ks   = (f >> 2) & 7;
  int nf   = f & 3;
  const float* w = mat ? w2 : w1;
  unsigned*    p = mat ? p2 : p1;
  int n  = (nq * 4 + nf) * 16 + (lane & 15);
  int k0 = ks * 32 + ((lane & 16) ? 16 : 0);
  unsigned* dst = p + ((size_t)(f * 32 + lane)) * 8;
#pragma unroll
  for (int j = 0; j < 8; ++j) {
    dst[j] = pack2(w[(size_t)(k0 + 2 * j) * OO + n],
                   w[(size_t)(k0 + 2 * j + 1) * OO + n]);
  }
}

// ---------------------------------------------------------------------------
// TDM: copy one contiguous 32KB region (8192 x 4B elements, 1 row) to LDS.
// D# per CDNA5 ISA:
//   g0: count=1 | lds_addr | global_addr[31:0] | global_addr[56:32]+type=2
//   g1: data_size=4B; tensor_dim0=8192; tensor_dim1=1; tile_dim0=8192;
//       tile_dim1=1; tensor_dim0_stride=8192
// Groups 2/3 zero (<=2D tensor: their fields are unused).
// ---------------------------------------------------------------------------
static __device__ __forceinline__ void tdm_load_32k(const void* gptr,
                                                    unsigned lds_off) {
  unsigned long long ga = (unsigned long long)gptr;
  tdm_g0_t g0;
  g0[0] = 1u;                                            // count=1
  g0[1] = lds_off;                                       // lds_addr (bytes)
  g0[2] = (unsigned)(ga & 0xffffffffu);                  // global_addr lo
  g0[3] = (unsigned)((ga >> 32) & 0x01ffffffu) | (2u << 30); // hi | type=2
  tdm_g1_t g1;
  g1[0] = (int)(2u << 16);                               // data_size=4B
  g1[1] = (int)(0x2000u << 16);                          // tensor_dim0 lo=8192
  g1[2] = (int)(1u << 16);                               // dim0 hi=0, dim1=1
  g1[3] = (int)(0x2000u << 16);                          // tile_dim0=8192
  g1[4] = 1;                                             // tile_dim1=1
  g1[5] = 0x2000;                                        // dim0_stride=8192
  g1[6] = 0;
  g1[7] = 0;
  tdm_g2_t z4 = {0, 0, 0, 0};
  tdm_g3_t z8 = {0, 0, 0, 0, 0, 0, 0, 0};
  __builtin_amdgcn_tensor_load_to_lds(g0, g1, z4, z4, z8, 0);
}

// ---------------------------------------------------------------------------
// Kernel 2: fused dual GEMM.  blockIdx.x = mb*4 + nq  (mb<625, nq<4).
// Wave w computes rows [mb*128 + w*16, +16) x cols [nq*64, +64).
// Per K-step: A fragment from global (fp32 -> bf16), all 8 B fragments
// (4 nf x 2 matrices) batch-loaded from LDS, then 8 back-to-back WMMAs.
// A order used here (contiguous k, same as B packing): lane<16 row=lane K{0..15};
//                            lane>=16 K{16..31}.
// C/D layout: VGPR r -> M = r + (lane>=16 ? 8 : 0), N = lane&15.
// ---------------------------------------------------------------------------
__global__ __launch_bounds__(256)
void gemm_dual(const float* __restrict__ x,        // [80000, 256]
               const unsigned* __restrict__ p1,    // packed W1 (128 KB)
               const unsigned* __restrict__ p2,    // packed W2 (128 KB)
               const float* __restrict__ bias,     // [256]
               float* __restrict__ out,            // [80000, 256]
               float* __restrict__ support) {      // [80000, 256]
  __shared__ unsigned char ldsB[65536];            // [mat(2)][32 KB region]

  const int mb    = blockIdx.x >> 2;
  const int nq    = blockIdx.x & 3;
  const int lane  = threadIdx.x & 31;
  const int w     = threadIdx.x >> 5;
  const int m     = lane & 15;
  const int hi    = (lane >> 4) & 1;
  const int rows0 = mb * 128 + w * 16;

  // Wave 0 issues both TDM copies of this N-quarter's fragment regions.
  if (threadIdx.x < 32) {
    const unsigned lds_base = (unsigned)(unsigned long long)(&ldsB[0]);
    tdm_load_32k((const void*)(p1 + (size_t)nq * 8192), lds_base);
    tdm_load_32k((const void*)(p2 + (size_t)nq * 8192), lds_base + 32768u);
    __builtin_amdgcn_s_wait_tensorcnt(0);
  }
  __syncthreads();

  v8f acc1[4], acc2[4];
#pragma unroll
  for (int nf = 0; nf < 4; ++nf) {
    acc1[nf] = (v8f){0.f, 0.f, 0.f, 0.f, 0.f, 0.f, 0.f, 0.f};
    acc2[nf] = (v8f){0.f, 0.f, 0.f, 0.f, 0.f, 0.f, 0.f, 0.f};
  }

  const float* arow = x + (size_t)(rows0 + m) * FF;

  for (int ks = 0; ks < 8; ++ks) {
    // ---- A fragment: two 16B-aligned fp32 chunk pairs -> bf16 ----
    const int kbase = ks * 32 + hi * 16;
    const float4* ap0 = (const float4*)(arow + kbase);
    const float4* ap1 = (const float4*)(arow + kbase + 8);
    float4 a0 = ap0[0], a1 = ap0[1];
    float4 a2 = ap1[0], a3 = ap1[1];
    union { v16bf v; __bf16 e[16]; } A;
    A.e[0]  = (__bf16)a0.x; A.e[1]  = (__bf16)a0.y;
    A.e[2]  = (__bf16)a0.z; A.e[3]  = (__bf16)a0.w;
    A.e[4]  = (__bf16)a1.x; A.e[5]  = (__bf16)a1.y;
    A.e[6]  = (__bf16)a1.z; A.e[7]  = (__bf16)a1.w;
    A.e[8]  = (__bf16)a2.x; A.e[9]  = (__bf16)a2.y;
    A.e[10] = (__bf16)a2.z; A.e[11] = (__bf16)a2.w;
    A.e[12] = (__bf16)a3.x; A.e[13] = (__bf16)a3.y;
    A.e[14] = (__bf16)a3.z; A.e[15] = (__bf16)a3.w;

    // ---- batch-load all 8 B fragments for this K-step from LDS ----
    union Bfrag { v16bf v; uint4 q[2]; };
    Bfrag Bf[8];   // [nf][mat]
#pragma unroll
    for (int nf = 0; nf < 4; ++nf) {
#pragma unroll
      for (int mat = 0; mat < 2; ++mat) {
        const unsigned off =
            (unsigned)mat * 32768u + (((unsigned)(ks * 4 + nf) * 32u +
                                       (unsigned)lane) * 32u);
        const uint4* q = (const uint4*)(&ldsB[off]);
        Bf[nf * 2 + mat].q[0] = q[0];
        Bf[nf * 2 + mat].q[1] = q[1];
      }
    }

    // ---- 8 back-to-back WMMAs ----
#pragma unroll
    for (int nf = 0; nf < 4; ++nf) {
      acc1[nf] = __builtin_amdgcn_wmma_f32_16x16x32_bf16(
          false, A.v, false, Bf[nf * 2 + 0].v, (short)0, acc1[nf], false, false);
      acc2[nf] = __builtin_amdgcn_wmma_f32_16x16x32_bf16(
          false, A.v, false, Bf[nf * 2 + 1].v, (short)0, acc2[nf], false, false);
    }
  }

  // ---- epilogue: out = x@W1 + bias ; support = x@W2 ----
#pragma unroll
  for (int nf = 0; nf < 4; ++nf) {
    const int col = nq * 64 + nf * 16 + m;
    const float bv = bias[col];
#pragma unroll
    for (int r = 0; r < 8; ++r) {
      const size_t idx = (size_t)(rows0 + r + hi * 8) * OO + col;
      out[idx]     = acc1[nf][r] + bv;
      support[idx] = acc2[nf][r];
    }
  }
}

// ---------------------------------------------------------------------------
// Kernel 3: edge scatter. One block per edge, lane = output channel.
// out[b, rows[e], o] += vals[e] * support[b, cols[e], o]  (L2 atomics)
// ---------------------------------------------------------------------------
__global__ __launch_bounds__(256)
void scatter_edges(const float* __restrict__ support,
                   const int* __restrict__ adj_rows,
                   const int* __restrict__ adj_cols,
                   const float* __restrict__ adj_vals,
                   float* __restrict__ out) {
  const int e = blockIdx.x;
  const int o = threadIdx.x;
  const int r = adj_rows[e];
  const int c = adj_cols[e];
  const float v = adj_vals[e];
  const size_t plane = (size_t)NN * OO;
#pragma unroll
  for (int b = 0; b < BB; ++b) {
    float msg = v * support[(size_t)b * plane + (size_t)c * OO + o];
    atomicAdd(out + (size_t)b * plane + (size_t)r * OO + o, msg);
  }
}

// ---------------------------------------------------------------------------
extern "C" void kernel_launch(void* const* d_in, const int* in_sizes, int n_in,
                              void* d_out, int out_size, void* d_ws, size_t ws_size,
                              hipStream_t stream) {
  const float* x    = (const float*)d_in[0];
  const int*   rows = (const int*)  d_in[1];
  const int*   cols = (const int*)  d_in[2];
  const float* vals = (const float*)d_in[3];
  const float* w1   = (const float*)d_in[4];
  const float* w2   = (const float*)d_in[5];
  const float* bias = (const float*)d_in[6];
  float* out = (float*)d_out;

  char* ws = (char*)d_ws;
  const size_t supBytes  = (size_t)MROWS * OO * sizeof(float);  // ~81.9 MB
  const size_t packBytes = (size_t)FF * OO * 2;                 // 128 KB each
  float*    support = (float*)ws;
  unsigned* p1 = (unsigned*)(ws + supBytes);
  unsigned* p2 = (unsigned*)(ws + supBytes + packBytes);

  pack_weights<<<32, 256, 0, stream>>>(w1, w2, p1, p2);
  gemm_dual<<<(MROWS / 128) * 4, 256, 0, stream>>>(x, p1, p2, bias, out, support);
  scatter_edges<<<EE, 256, 0, stream>>>(support, rows, cols, vals, out);
}
